// AttentionHead_40458591928781
// MI455X (gfx1250) — matrix operand order repaired and checked by reading, not compiled, not run on hardware
//
#include <hip/hip_runtime.h>
#include <math.h>

typedef _Float16 f16;
typedef __attribute__((ext_vector_type(16))) _Float16 v16h;
typedef __attribute__((ext_vector_type(8)))  _Float16 v8h;
typedef __attribute__((ext_vector_type(8)))  float    v8f;

#define D_MODEL 1024
#define HS      64
#define BB      4
#define TT      2048
#define NROWS   (BB*TT)        // 8192
#define SCALE   8.0f           // sqrt(hs) per reference (faithful to source)
#define KPH     128            // K-dim per LDS staging phase
#define LDPAD   8              // 16B row pad -> conflict-free column reads

static __device__ __forceinline__ v8f wmma_f16(v16h a, v16h b, v8f c) {
  // D = A(16x32 f16) * B(32x16 f16) + C(16x16 f32)
  return __builtin_amdgcn_wmma_f32_16x16x32_f16(false, a, false, b, (short)0, c,
                                                false, false);
}

// ---------------------------------------------------------------------------
// Kernel 1: convert x -> f16, weights -> f16 transposed ([64][1024])
// ---------------------------------------------------------------------------
__global__ void cvt_kernel(const float* __restrict__ x,
                           const float* __restrict__ Wq,
                           const float* __restrict__ Wk,
                           const float* __restrict__ Wv,
                           f16* __restrict__ xh,
                           f16* __restrict__ Wqt,
                           f16* __restrict__ Wkt,
                           f16* __restrict__ Wvt) {
  int tid = blockIdx.x * blockDim.x + threadIdx.x;
  int stride = gridDim.x * blockDim.x;
  for (int i = tid; i < NROWS * D_MODEL; i += stride) xh[i] = (f16)x[i];
  for (int i = tid; i < HS * D_MODEL; i += stride) {
    int col = i / D_MODEL;          // head dim
    int k   = i - col * D_MODEL;    // reduction dim
    Wqt[i] = (f16)Wq[k * HS + col];
    Wkt[i] = (f16)Wk[k * HS + col];
    Wvt[i] = (f16)Wv[k * HS + col];
  }
}

// ---------------------------------------------------------------------------
// Kernel 2: fused QKV projection with LDS-staged weights. 8 waves per block
// cooperatively stage Wq/Wk/Wv (transposed f16) into LDS in 128-wide K
// phases; B-fragment reads become ~30-cycle conflict-free ds_load_b128
// instead of L2 round-trips, and ~51KB LDS/block leaves room for multiple
// blocks per WGP so residual stalls are hidden by occupancy.
// ---------------------------------------------------------------------------
__global__ __launch_bounds__(256) void proj_kernel(
    const f16* __restrict__ xh,
    const f16* __restrict__ Wqt, const f16* __restrict__ Wkt,
    const f16* __restrict__ Wvt,
    const float* __restrict__ bq, const float* __restrict__ bk,
    f16* __restrict__ Qh, f16* __restrict__ Kh, f16* __restrict__ Vt) {
  __shared__ __align__(16) f16 smem[3][HS][KPH + LDPAD];

  const int tid  = threadIdx.x;
  const int lane = tid & 31;
  const int half = lane >> 4;
  const int l15  = lane & 15;
  const int wloc = tid >> 5;                         // 0..7
  const int row0 = (blockIdx.x * 8 + wloc) * 16;     // 16-row tile
  const f16* xrow = xh + (size_t)(row0 + l15) * D_MODEL;

  v8f aq[4], ak[4], av[4];
#pragma unroll
  for (int nt = 0; nt < 4; ++nt) { aq[nt] = (v8f){}; ak[nt] = (v8f){}; av[nt] = (v8f){}; }

  const f16* __restrict__ Wt0 = Wqt;
  const f16* __restrict__ Wt1 = Wkt;
  const f16* __restrict__ Wt2 = Wvt;

  for (int p = 0; p < D_MODEL / KPH; ++p) {
    // ---- stage this K-phase of all three weight matrices into LDS
    __syncthreads();                                  // prior phase fully read
#pragma unroll
    for (int j = 0; j < 4; ++j) {
      const int c   = tid + j * 256;                  // 1024 16B-chunks/matrix
      const int col = c >> 4;                         // 16 chunks per column
      const int k   = (c & 15) * 8;
      const size_t gofs = (size_t)col * D_MODEL + p * KPH + k;
      *(v8h*)&smem[0][col][k] = *(const v8h*)(Wt0 + gofs);
      *(v8h*)&smem[1][col][k] = *(const v8h*)(Wt1 + gofs);
      *(v8h*)&smem[2][col][k] = *(const v8h*)(Wt2 + gofs);
    }
    __syncthreads();
    // prefetch next phase's x rows while computing this phase
    __builtin_prefetch(xrow + p * KPH + KPH, 0, 3);

    // ---- 4 K-steps of 32 against LDS-resident weights
#pragma unroll
    for (int ks = 0; ks < KPH; ks += 32) {
      const int kk = p * KPH + ks;
      union { v16h v; v8h h[2]; } a;
      const f16* arow = xrow + kk + half * 8;
      a.h[0] = *(const v8h*)(arow);
      a.h[1] = *(const v8h*)(arow + 16);
#pragma unroll
      for (int nt = 0; nt < 4; ++nt) {
        const int col = nt * 16 + l15;
        union { v16h v; v8h h[2]; } fq, fk, fv;
        const f16* b0 = &smem[0][col][ks + half * 8];
        const f16* b1 = &smem[1][col][ks + half * 8];
        const f16* b2 = &smem[2][col][ks + half * 8];
        fq.h[0] = *(const v8h*)(b0); fq.h[1] = *(const v8h*)(b0 + 16);
        fk.h[0] = *(const v8h*)(b1); fk.h[1] = *(const v8h*)(b1 + 16);
        fv.h[0] = *(const v8h*)(b2); fv.h[1] = *(const v8h*)(b2 + 16);
        aq[nt] = wmma_f16(a.v, fq.v, aq[nt]);
        ak[nt] = wmma_f16(a.v, fk.v, ak[nt]);
        av[nt] = wmma_f16(a.v, fv.v, av[nt]);
      }
    }
  }

#pragma unroll
  for (int nt = 0; nt < 4; ++nt) {
    const int col = nt * 16 + l15;
    const float biasq = bq[col];
    const float biask = bk[col];
#pragma unroll
    for (int r = 0; r < 8; ++r) {
      const int row = row0 + r + half * 8;           // global row in [0,8192)
      Qh[(size_t)row * HS + col] = (f16)(aq[nt][r] + biasq);
      Kh[(size_t)row * HS + col] = (f16)(ak[nt][r] + biask);
      const int b = row >> 11, t = row & (TT - 1);
      Vt[((size_t)b * HS + col) * TT + t] = (f16)av[nt][r];
    }
  }
}

// ---------------------------------------------------------------------------
// Kernel 3: causal flash attention. One wave per (batch, 16-row query tile).
// 32 keys per step: QK^T (4 wmma) -> online softmax -> P@V (4 wmma).
// K and V fragments are loaded together at the top of the step so the V
// loads overlap the softmax VALU work; next step is prefetched into WGP$.
// ---------------------------------------------------------------------------
__global__ __launch_bounds__(128)
__attribute__((amdgpu_waves_per_eu(1))) void attn_kernel(
    const f16* __restrict__ Qh, const f16* __restrict__ Kh,
    const f16* __restrict__ Vt, float* __restrict__ out) {
  __shared__ __align__(32) f16 ptile[4][16 * 32];   // per-wave P staging

  const int lane   = threadIdx.x & 31;
  const int wlocal = threadIdx.x >> 5;
  const int wave   = blockIdx.x * 4 + wlocal;       // 0..511
  const int b      = wave >> 7;                     // 128 q-tiles / batch
  const int qt     = wave & 127;
  const int half   = lane >> 4;
  const int l15    = lane & 15;
  const int qbase  = qt * 16;

  const f16* Qb = Qh + (size_t)b * TT * HS;
  const f16* Kb = Kh + (size_t)b * TT * HS;
  const f16* Vb = Vt + (size_t)b * HS * TT;

  // Q A-fragments for the two 32-wide K-steps of the hs=64 reduction
  union { v16h v; v8h h[2]; } aq0, aq1;
  const f16* qrow = Qb + (size_t)(qbase + l15) * HS;
  aq0.h[0] = *(const v8h*)(qrow + half * 16);
  aq0.h[1] = *(const v8h*)(qrow + half * 16 + 8);
  aq1.h[0] = *(const v8h*)(qrow + 32 + half * 16);
  aq1.h[1] = *(const v8h*)(qrow + 32 + half * 16 + 8);

  v8f acc[4];
#pragma unroll
  for (int nt = 0; nt < 4; ++nt) acc[nt] = (v8f){};
  float m_[8], l_[8];
#pragma unroll
  for (int r = 0; r < 8; ++r) { m_[r] = -INFINITY; l_[r] = 0.0f; }

  f16* lds = ptile[wlocal];
  const int kend = qbase + 16;                      // causal: keys < kend

  for (int kb = 0; kb < kend; kb += 32) {
    // ---- load all K and V fragments for this 32-key step (one clause)
    const f16* kr0 = Kb + (size_t)(kb + l15) * HS;
    const f16* kr1 = Kb + (size_t)(kb + 16 + l15) * HS;
    v16h bk0 = *(const v16h*)(kr0 + half * 16);
    v16h bk1 = *(const v16h*)(kr0 + 32 + half * 16);
    v16h bk2 = *(const v16h*)(kr1 + half * 16);
    v16h bk3 = *(const v16h*)(kr1 + 32 + half * 16);
    v16h bv[4];
#pragma unroll
    for (int nt = 0; nt < 4; ++nt)
      bv[nt] = *(const v16h*)(Vb + (size_t)(nt * 16 + l15) * TT + kb + half * 16);

    // ---- prefetch next step's K/V into the near cache (global_prefetch_b8)
    if (kb + 32 < kend) {
      __builtin_prefetch(Kb + (size_t)(kb + 32 + l15) * HS, 0, 3);
      __builtin_prefetch(Vb + (size_t)l15 * TT + kb + 32, 0, 3);
    }

    // ---- S = Q K^T for 2 sub-tiles of 16 keys
    v8f s0 = (v8f){}, s1 = (v8f){};
    s0 = wmma_f16(aq0.v, bk0, s0);
    s0 = wmma_f16(aq1.v, bk1, s0);
    s1 = wmma_f16(aq0.v, bk2, s1);
    s1 = wmma_f16(aq1.v, bk3, s1);

    // ---- scale + causal mask (wave-uniform guard; lanewise select)
    const bool mask0 = (kb + 15      > qbase);
    const bool mask1 = (kb + 16 + 15 > qbase);
#pragma unroll
    for (int r = 0; r < 8; ++r) {
      const int row = qbase + r + half * 8;
      float v0 = s0[r] * SCALE;
      float v1 = s1[r] * SCALE;
      if (mask0) v0 = (kb + l15      > row) ? -3.0e38f : v0;
      if (mask1) v1 = (kb + 16 + l15 > row) ? -3.0e38f : v1;
      s0[r] = v0; s1[r] = v1;
    }
    // ---- online softmax (rows live in 16-lane halves: xor 1,2,4,8)
#pragma unroll
    for (int r = 0; r < 8; ++r) {
      float t = fmaxf(s0[r], s1[r]);
      t = fmaxf(t, __shfl_xor(t, 1));
      t = fmaxf(t, __shfl_xor(t, 2));
      t = fmaxf(t, __shfl_xor(t, 4));
      t = fmaxf(t, __shfl_xor(t, 8));
      const float newm  = fmaxf(m_[r], t);
      const float alpha = __expf(m_[r] - newm);
      const float p0 = __expf(s0[r] - newm);
      const float p1 = __expf(s1[r] - newm);
      float rs = p0 + p1;
      rs += __shfl_xor(rs, 1);
      rs += __shfl_xor(rs, 2);
      rs += __shfl_xor(rs, 4);
      rs += __shfl_xor(rs, 8);
      l_[r] = l_[r] * alpha + rs;
      m_[r] = newm;
      acc[0][r] *= alpha; acc[1][r] *= alpha;
      acc[2][r] *= alpha; acc[3][r] *= alpha;
      // stage P into LDS in row-major [16][32] f16
      const int lrow = r + half * 8;
      lds[lrow * 32 + l15]      = (f16)p0;
      lds[lrow * 32 + 16 + l15] = (f16)p1;
    }
    // LDS from same wave is issue-ordered; wait for DS completion before
    // the dependent A-fragment loads (CDNA5 split counters).
    asm volatile("s_wait_dscnt 0" ::: "memory");

    // ---- reload P as A-fragment (16x32 f16 layout)
    union { v16h v; v8h h[2]; } ap;
    const f16* prow = lds + l15 * 32 + half * 16;
    ap.h[0] = *(const v8h*)(prow);
    ap.h[1] = *(const v8h*)(prow + 8);

    // ---- O += P @ V   (V fragments already in registers)
#pragma unroll
    for (int nt = 0; nt < 4; ++nt) {
      acc[nt] = wmma_f16(ap.v, bv[nt], acc[nt]);
    }
  }

  // ---- normalize and store f32 output [B,T,64]
#pragma unroll
  for (int r = 0; r < 8; ++r) {
    const float inv = 1.0f / l_[r];
    const int row = qbase + r + half * 8;
#pragma unroll
    for (int nt = 0; nt < 4; ++nt) {
      out[((size_t)b * TT + row) * HS + nt * 16 + l15] = acc[nt][r] * inv;
    }
  }
}

// ---------------------------------------------------------------------------
extern "C" void kernel_launch(void* const* d_in, const int* in_sizes, int n_in,
                              void* d_out, int out_size, void* d_ws, size_t ws_size,
                              hipStream_t stream) {
  const float* x  = (const float*)d_in[0];
  const float* Wq = (const float*)d_in[1];
  const float* bq = (const float*)d_in[2];
  const float* Wk = (const float*)d_in[3];
  const float* bk = (const float*)d_in[4];
  const float* Wv = (const float*)d_in[5];
  float* out = (float*)d_out;

  // workspace layout (bytes)
  char* ws = (char*)d_ws;
  f16* xh  = (f16*)(ws);                               // 8192*1024*2 = 16 MiB
  f16* Wqt = (f16*)(ws + 16777216);                    // 64*1024*2
  f16* Wkt = (f16*)(ws + 16777216 + 131072);
  f16* Wvt = (f16*)(ws + 16777216 + 262144);
  f16* Qh  = (f16*)(ws + 16777216 + 393216);           // 8192*64*2 = 1 MiB
  f16* Kh  = (f16*)(ws + 16777216 + 393216 + 1048576);
  f16* Vt  = (f16*)(ws + 16777216 + 393216 + 2097152); // transposed [B][64][T]

  cvt_kernel<<<1024, 256, 0, stream>>>(x, Wq, Wk, Wv, xh, Wqt, Wkt, Wvt);
  proj_kernel<<<64, 256, 0, stream>>>(xh, Wqt, Wkt, Wvt, bq, bk, Qh, Kh, Vt);
  attn_kernel<<<128, 128, 0, stream>>>(Qh, Kh, Vt, out);
}
